// SLOFDetector_41721312313544
// MI455X (gfx1250) — matrix operand order repaired and checked by reading, not compiled, not run on hardware
//
#include <hip/hip_runtime.h>
#include <math.h>

#define NPTS 8192
#define DIM  256
#define KNN  32
#define KP1  33    // need 33 smallest (excluding self)
#define CSTRIDE 260  // padded chunk-row stride (bank-conflict-free C-tile stores)

typedef __attribute__((ext_vector_type(16))) _Float16 v16h;
typedef __attribute__((ext_vector_type(8)))  _Float16 v8h;
typedef __attribute__((ext_vector_type(4)))  _Float16 v4h;
typedef __attribute__((ext_vector_type(8)))  float    v8f;

// ---------------- kernel 0a: f32 -> f16 conversion (vectorized x4) -------------
__global__ void slof_conv_kernel(const float* __restrict__ feat,
                                 _Float16* __restrict__ feat16) {
  int idx = blockIdx.x * blockDim.x + threadIdx.x;
  if (idx < NPTS * DIM / 4) {
    float4 v = ((const float4*)feat)[idx];
    v4h h;
    h[0] = (_Float16)v.x; h[1] = (_Float16)v.y;
    h[2] = (_Float16)v.z; h[3] = (_Float16)v.w;
    ((v4h*)feat16)[idx] = h;
  }
}

// ---------------- kernel 0b: squared row norms (f32, exact) --------------------
__global__ void slof_norms_kernel(const float* __restrict__ feat,
                                  float* __restrict__ x2) {
  int row = blockIdx.x * blockDim.x + threadIdx.x;
  if (row < NPTS) {
    float s = 0.0f;
    #pragma unroll 8
    for (int d = 0; d < DIM; ++d) {
      float v = feat[row * DIM + d];
      s = fmaf(v, v, s);
    }
    x2[row] = s;
  }
}

// ---- compute one 16x256 distance chunk via WMMA (4 tiles per wave) ------------
__device__ __forceinline__ void compute_chunk(
    const _Float16* __restrict__ feat16, const float* __restrict__ x2,
    const float* x2row, float* distBuf, const v16h aFrag[8],
    int i0, int j0chunk, int wid, int lane) {
  const int colN = lane & 15;        // N within tile
  const int mOff = (lane >> 4) * 8;  // C/D: lanes 16-31 hold M = r+8
  const int kb   = (lane >> 4) * 16; // B: lanes 16-31 hold K = 16..31
  #pragma unroll
  for (int t = 0; t < 4; ++t) {
    const int ct = wid * 4 + t;
    const int j0 = j0chunk + ct * 16;
    const int bRow = j0 + colN;      // column point index for this lane
    v8f acc = {};
    #pragma unroll
    for (int ks = 0; ks < 8; ++ks) {
      const v16h b = *(const v16h*)(feat16 + (size_t)bRow * DIM + ks * 32 + kb);
      acc = __builtin_amdgcn_wmma_f32_16x16x32_f16(
          false, aFrag[ks], false, b, (short)0, acc, false, false);
    }
    const float x2c = x2[bRow];
    #pragma unroll
    for (int r = 0; r < 8; ++r) {
      const int m  = r + mOff;
      const int gi = i0 + m;
      const int gj = j0 + colN;
      float d2 = x2row[m] + x2c - 2.0f * acc[r];
      float dd = sqrtf(fmaxf(d2, 0.0f));
      if (gj == gi) dd = __builtin_huge_valf();   // exclude self
      distBuf[m * CSTRIDE + ct * 16 + colN] = dd;
    }
  }
}

// ---- wave-parallel top-33 update: ballot filter + lane-0 sorted insertion -----
__device__ __forceinline__ void select_chunk(
    const float* distBuf, float* topVal, int* topIdx,
    int i0, int j0chunk, int wid, int lane) {
  #pragma unroll
  for (int rr = 0; rr < 4; ++rr) {
    const int row = wid * 4 + rr;    // each wave owns 4 rows
    const int gi  = i0 + row;
    float th = topVal[row * KP1 + 32];
    for (int s = 0; s < 8; ++s) {
      const int q  = s * 32 + lane;
      const float dd = distBuf[row * CSTRIDE + q];
      unsigned int m = __builtin_amdgcn_ballot_w32(dd < th);
      if (m) {                        // uniform branch
        if (lane == 0) {
          while (m) {
            const int src = __builtin_ctz(m);
            m &= m - 1;
            const int qq = s * 32 + src;
            const float v = distBuf[row * CSTRIDE + qq];
            if (v < th) {             // re-check against tightened threshold
              const int gj = j0chunk + qq;
              const int p  = gj - (gj > gi ? 1 : 0);  // raw diag-removed index
              int pos = 32;
              while (pos > 0 && topVal[row * KP1 + pos - 1] > v) {
                topVal[row * KP1 + pos] = topVal[row * KP1 + pos - 1];
                topIdx[row * KP1 + pos] = topIdx[row * KP1 + pos - 1];
                --pos;
              }
              topVal[row * KP1 + pos] = v;
              topIdx[row * KP1 + pos] = p;
              th = topVal[row * KP1 + 32];
            }
          }
        }
        th = __shfl(th, 0, 32);       // broadcast tightened threshold
      }
    }
  }
}

// ---------------- kernel 1: pipelined distances + streaming top-33 -------------
__global__ void __launch_bounds__(128)
slof_main_kernel(const _Float16* __restrict__ feat16,
                 const float* __restrict__ x2,
                 float* __restrict__ kth,       // a[i,32]
                 float* __restrict__ dkf,       // a_full_d[i,32] = 32nd smallest, no self
                 int*   __restrict__ knn) {     // raw diag-removed indices of 32-NN
  __shared__ float distBuf[2][16 * CSTRIDE];
  __shared__ float topVal[16 * KP1];
  __shared__ int   topIdx[16 * KP1];
  __shared__ float x2row[16];

  const int tid  = threadIdx.x;
  const int wid  = tid >> 5;
  const int lane = tid & 31;
  const int i0   = blockIdx.x * 16;

  for (int t = tid; t < 16 * KP1; t += 128) {
    topVal[t] = __builtin_huge_valf();
    topIdx[t] = 0;
  }
  if (tid < 16) x2row[tid] = x2[i0 + tid];
  __syncthreads();

  // preload A fragments for all of D (contiguous k order, same as B):
  // lanes 0-15: M=lane, K 0..15; lanes 16-31: M=lane-16, K 16..31
  const int aRow = i0 + (lane & 15);
  const int koff = (lane >> 4) * 16;
  v16h aFrag[8];
  #pragma unroll
  for (int ks = 0; ks < 8; ++ks) {
    const _Float16* ap = feat16 + (size_t)aRow * DIM + ks * 32 + koff;
    v8h lo = *(const v8h*)(ap);        // K = base+0..7
    v8h hi = *(const v8h*)(ap + 8);    // K = base+8..15
    aFrag[ks] = __builtin_shufflevector(lo, hi,
        0, 1, 2, 3, 4, 5, 6, 7, 8, 9, 10, 11, 12, 13, 14, 15);
  }

  // software pipeline: compute chunk c+1 while selecting chunk c
  compute_chunk(feat16, x2, x2row, distBuf[0], aFrag, i0, 0, wid, lane);
  __syncthreads();
  for (int c = 0; c < 32; ++c) {
    if (c + 1 < 32)
      compute_chunk(feat16, x2, x2row, distBuf[(c + 1) & 1], aFrag,
                    i0, (c + 1) * 256, wid, lane);
    select_chunk(distBuf[c & 1], topVal, topIdx, i0, c * 256, wid, lane);
    __syncthreads();
  }

  if (tid < 16) {
    const int gi = i0 + tid;
    kth[gi] = topVal[tid * KP1 + 32];   // a[i, k]       (k=32)
    dkf[gi] = topVal[tid * KP1 + 31];   // a_full_d[i,k] (self 0 shifts by one)
    for (int j = 0; j < KNN; ++j)
      knn[gi * KNN + j] = topIdx[tid * KP1 + j];
  }
}

// ---------------- kernel 2: final score ---------------------------------------
__global__ void slof_finish_kernel(const float* __restrict__ kth,
                                   const float* __restrict__ dkf,
                                   const int*   __restrict__ knn,
                                   float* __restrict__ out) {
  int i = blockIdx.x * blockDim.x + threadIdx.x;
  if (i >= NPTS) return;
  float a = kth[i];
  float s = 0.0f;
  #pragma unroll
  for (int j = 0; j < KNN; ++j) {
    int p = knn[i * KNN + j];          // reference uses raw index (no diag fixup)
    s += a / dkf[p];
  }
  s *= (1.0f / (float)KNN);
  if (__builtin_isnan(s))      s = 1000.0f;
  else if (__builtin_isinf(s)) s = (s > 0.0f) ? 1000.0f : 0.0f;
  out[i] = s;
}

extern "C" void kernel_launch(void* const* d_in, const int* in_sizes, int n_in,
                              void* d_out, int out_size, void* d_ws, size_t ws_size,
                              hipStream_t stream) {
  (void)in_sizes; (void)n_in; (void)out_size; (void)ws_size;
  const float* feat = (const float*)d_in[0];
  float* out = (float*)d_out;

  char* ws = (char*)d_ws;
  _Float16* feat16 = (_Float16*)ws;                          // 4 MB
  float* x2  = (float*)(ws + (size_t)NPTS * DIM * 2);        // 32 KB
  float* kth = x2 + NPTS;                                    // 32 KB
  float* dkf = kth + NPTS;                                   // 32 KB
  int*   knn = (int*)(dkf + NPTS);                           // 1 MB

  slof_conv_kernel <<<(NPTS * DIM / 4 + 255) / 256, 256, 0, stream>>>(feat, feat16);
  slof_norms_kernel<<<(NPTS + 255) / 256,        256, 0, stream>>>(feat, x2);
  slof_main_kernel <<<NPTS / 16,                 128, 0, stream>>>(feat16, x2, kth, dkf, knn);
  slof_finish_kernel<<<(NPTS + 255) / 256,       256, 0, stream>>>(kth, dkf, knn, out);
}
